// Jointer_19963007992158
// MI455X (gfx1250) — matrix operand order repaired and checked by reading, not compiled, not run on hardware
//
#include <hip/hip_runtime.h>
#include <hip/hip_fp16.h>

// Problem shape (fixed by the reference): B=8, S=T=2048, D=128.
#define BATCH 8
#define SDIM  2048
#define TDIM  2048
#define DDIM  128

typedef __attribute__((ext_vector_type(16))) _Float16 v16h;
typedef __attribute__((ext_vector_type(8)))  float    v8f;
typedef __attribute__((ext_vector_type(4)))  _Float16 h4;

union FragH {          // 16x32 f16 A-fragment / 32x16 f16 B-fragment: 8 VGPRs
    v16h h;
    uint4 q[2];        // two 16-byte chunks per lane
};

// gfx1250 async direct-to-LDS staging (GLOBAL_LOAD_ASYNC_TO_LDS_B128,
// tracked by ASYNCcnt). Probe round 2 showed the builtin's param types:
// arg0 = int4* in addrspace(1) (global), arg1 = LDS side, then imm offset,
// imm cpol.
#if __has_builtin(__builtin_amdgcn_global_load_async_to_lds_b128) && \
    __has_builtin(__builtin_amdgcn_s_wait_asynccnt)
#define USE_ASYNC_LDS 1
typedef int v4i_vs __attribute__((vector_size(16)));
typedef __attribute__((address_space(1))) v4i_vs* gvec_ptr;
typedef __attribute__((address_space(3))) v4i_vs* lvec_ptr;
#else
#define USE_ASYNC_LDS 0
#endif

// ---------------------------------------------------------------------------
// Phase 1: L2-normalize each row, fold in the row mask (masked rows -> 0),
// convert to f16. One wave32 per 128-float row; 4 floats per lane.
// ---------------------------------------------------------------------------
__global__ __launch_bounds__(256) void normalize_rows(
    const float* __restrict__ src, const float* __restrict__ tgt,
    const unsigned char* __restrict__ mask_s,
    const unsigned char* __restrict__ mask_t,
    _Float16* __restrict__ outS, _Float16* __restrict__ outT)
{
    const int wave = threadIdx.x >> 5;
    const int lane = threadIdx.x & 31;
    const long row = (long)blockIdx.x * 8 + wave;   // 32768 rows total

    const long nS = (long)BATCH * SDIM;
    const float* in;
    _Float16*    out;
    int          m;
    if (row < nS) {
        in  = src + row * DDIM;
        out = outS + row * DDIM;
        m   = mask_s[row];
    } else {
        const long r = row - nS;
        in  = tgt + r * DDIM;
        out = outT + r * DDIM;
        m   = mask_t[r];
    }

    const float4 v = ((const float4*)in)[lane];
    float ss = v.x * v.x + v.y * v.y + v.z * v.z + v.w * v.w;
    #pragma unroll
    for (int off = 16; off > 0; off >>= 1)
        ss += __shfl_xor(ss, off, 32);

    float scale = 1.0f / fmaxf(sqrtf(ss), 1e-12f);  // matches F.normalize eps
    if (!m) scale = 0.0f;                            // fold mask: row -> 0

    h4 h;
    h.x = (_Float16)(v.x * scale);
    h.y = (_Float16)(v.y * scale);
    h.z = (_Float16)(v.z * scale);
    h.w = (_Float16)(v.w * scale);
    ((h4*)out)[lane] = h;
}

// ---------------------------------------------------------------------------
// Phase 2: batched NT GEMM  C[b] = relu(Sn[b] (2048x128) * Tn[b]^T) in fp32,
// via v_wmma_f32_16x16x32_f16. Block = 256 threads = 8 waves, C tile 128x64.
// A tile 128x128 f16 + B tile 64x128 f16 staged in LDS (row stride padded to
// 136 halves = 272 B to decorrelate banks). Each wave: 32x32 C sub-tile.
// ---------------------------------------------------------------------------
#define TM  128
#define TN  64
#define LDA 136   // halves (128 + 8 pad)
#define LDB 136

__global__ __launch_bounds__(256) void gemm_relu(
    const _Float16* __restrict__ Sn, const _Float16* __restrict__ Tn,
    float* __restrict__ out)
{
    __shared__ _Float16 sA[TM * LDA];   // 34816 B
    __shared__ _Float16 sB[TN * LDB];   // 17408 B  (total 52224 B < 64 KB)

    const int b  = blockIdx.z;
    const int m0 = blockIdx.y * TM;
    const int n0 = blockIdx.x * TN;
    const int tid = threadIdx.x;

    const _Float16* Ab = Sn + ((long)b * SDIM + m0) * DDIM;
    const _Float16* Bb = Tn + ((long)b * TDIM + n0) * DDIM;

    // Stage tiles: rows are 128 halves = 16 chunks of 16 B.
#if USE_ASYNC_LDS
    // Async direct-to-LDS: no VGPR round trip; each wave waits on its own
    // ASYNCcnt, then the barrier makes all waves' tiles visible.
    #pragma unroll
    for (int c = tid; c < TM * 16; c += 256) {
        const int r = c >> 4, col = (c & 15) << 3;
        __builtin_amdgcn_global_load_async_to_lds_b128(
            (gvec_ptr)(Ab + (long)r * DDIM + col),
            (lvec_ptr)(&sA[r * LDA + col]), 0, 0);
    }
    #pragma unroll
    for (int c = tid; c < TN * 16; c += 256) {
        const int r = c >> 4, col = (c & 15) << 3;
        __builtin_amdgcn_global_load_async_to_lds_b128(
            (gvec_ptr)(Bb + (long)r * DDIM + col),
            (lvec_ptr)(&sB[r * LDB + col]), 0, 0);
    }
    __builtin_amdgcn_s_wait_asynccnt(0);
#else
    #pragma unroll
    for (int c = tid; c < TM * 16; c += 256) {
        const int r = c >> 4, col = (c & 15) << 3;
        *(uint4*)(&sA[r * LDA + col]) = *(const uint4*)(Ab + (long)r * DDIM + col);
    }
    #pragma unroll
    for (int c = tid; c < TN * 16; c += 256) {
        const int r = c >> 4, col = (c & 15) << 3;
        *(uint4*)(&sB[r * LDB + col]) = *(const uint4*)(Bb + (long)r * DDIM + col);
    }
#endif
    __syncthreads();

    const int wave = tid >> 5;
    const int lane = tid & 31;
    const int lh   = lane & 15;    // lane within half-wave
    const int hi   = lane >> 4;    // which half-wave
    const int mb   = (wave & 3) * 32;   // wave's M strip in tile
    const int nb   = (wave >> 2) * 32;  // wave's N strip in tile

    v8f acc[2][2] = {};   // [mt][nt] 16x16 fp32 accumulators

    #pragma unroll
    for (int ks = 0; ks < 4; ++ks) {
        const int k0 = ks * 32;

        // A fragment (ISA 7.12.2, 16-bit A 16x32):
        //   lanes 0-15:  row M=lh, K = k0+[0..7] and k0+[16..23]
        //   lanes 16-31: row M=lh, K = k0+[8..15] and k0+[24..31]
        FragH a[2];
        #pragma unroll
        for (int mt = 0; mt < 2; ++mt) {
            const int r  = mb + mt * 16 + lh;
            const int ao = k0 + hi * 8;
            a[mt].q[0] = *(const uint4*)(&sA[r * LDA + ao]);
            a[mt].q[1] = *(const uint4*)(&sA[r * LDA + ao + 16]);
        }

        // B fragment (32x16, K x N), fed from row-major Tn (NT GEMM):
        //   lanes 0-15:  col N=lh, K = k0+[0..7] and k0+[16..23]
        //   lanes 16-31: col N=lh, K = k0+[8..15] and k0+[24..31]
        FragH bf[2];
        #pragma unroll
        for (int nt = 0; nt < 2; ++nt) {
            const int r  = nb + nt * 16 + lh;
            const int bo = k0 + hi * 8;
            bf[nt].q[0] = *(const uint4*)(&sB[r * LDB + bo]);
            bf[nt].q[1] = *(const uint4*)(&sB[r * LDB + bo + 16]);
        }

        #pragma unroll
        for (int mt = 0; mt < 2; ++mt)
            #pragma unroll
            for (int nt = 0; nt < 2; ++nt)
                acc[mt][nt] = __builtin_amdgcn_wmma_f32_16x16x32_f16(
                    false, a[mt].h, false, bf[nt].h,
                    (short)0, acc[mt][nt], false, false);
    }

    // Epilogue: C/D layout -> VGPR v holds (M = 8*hi + v, N = lh). ReLU+store.
    float* C = out + ((long)b * SDIM + m0) * TDIM + n0;
    #pragma unroll
    for (int mt = 0; mt < 2; ++mt)
        #pragma unroll
        for (int nt = 0; nt < 2; ++nt)
            #pragma unroll
            for (int v = 0; v < 8; ++v) {
                const int m = mb + mt * 16 + hi * 8 + v;
                const int n = nb + nt * 16 + lh;
                const float x = acc[mt][nt][v];
                C[(long)m * TDIM + n] = x > 0.0f ? x : 0.0f;
            }
}

// ---------------------------------------------------------------------------
extern "C" void kernel_launch(void* const* d_in, const int* in_sizes, int n_in,
                              void* d_out, int out_size, void* d_ws, size_t ws_size,
                              hipStream_t stream)
{
    const float* src = (const float*)d_in[0];                 // [8,2048,128] f32
    const float* tgt = (const float*)d_in[1];                 // [8,2048,128] f32
    const unsigned char* ms = (const unsigned char*)d_in[2];  // [8,2048] bool
    const unsigned char* mt = (const unsigned char*)d_in[3];  // [8,2048] bool
    float* out = (float*)d_out;                               // [8,2048,2048] f32

    _Float16* nS = (_Float16*)d_ws;                     // 4 MB
    _Float16* nT = nS + (size_t)BATCH * SDIM * DDIM;    // 4 MB

    // Phase 1: 32768 rows, 8 rows (waves) per 256-thread block.
    const int rows = BATCH * (SDIM + TDIM);
    normalize_rows<<<rows / 8, 256, 0, stream>>>(src, tgt, ms, mt, nS, nT);

    // Phase 2: batched WMMA GEMM + ReLU.
    dim3 grid(TDIM / TN, SDIM / TM, BATCH);
    gemm_relu<<<grid, 256, 0, stream>>>(nS, nT, out);
}
